// PointerGeneratorDecoder_30923764531844
// MI455X (gfx1250) — matrix operand order repaired and checked by reading, not compiled, not run on hardware
//
#include <hip/hip_runtime.h>
#include <hip/hip_bf16.h>
#include <math.h>

// ---------------------------------------------------------------------------
// PointerGeneratorDecoder on MI455X (gfx1250, wave32, WMMA).
// GEMM: all-f16 operands (pre-converted / pre-transposed), f32 accumulate via
// v_wmma_f32_16x16x32_f16, 2x2 register tiling (32x32 C per wave),
// 16B vector loads for A, strided 2-byte loads for B in the inner loop.
// ---------------------------------------------------------------------------

#define BATCH 4
#define TLEN  256
#define SLEN  512
#define DMODEL 512
#define NHEAD 8
#define DHEAD 64
#define NLAYER 6
#define DFF   2048
#define VOCAB 32000
#define BT    (BATCH * TLEN)     // 1024
#define BS    (BATCH * SLEN)     // 2048

typedef __attribute__((ext_vector_type(16))) _Float16 v16h;
typedef __attribute__((ext_vector_type(8)))  _Float16 v8h;
typedef __attribute__((ext_vector_type(8)))  float    v8f;

// ------------------------------ reductions --------------------------------
__device__ __forceinline__ float block_sum(float v, float* sb) {
    int tid = threadIdx.x;
    sb[tid] = v; __syncthreads();
    for (int s = blockDim.x >> 1; s > 0; s >>= 1) {
        if (tid < s) sb[tid] += sb[tid + s];
        __syncthreads();
    }
    float r = sb[0]; __syncthreads();
    return r;
}
__device__ __forceinline__ float block_max(float v, float* sb) {
    int tid = threadIdx.x;
    sb[tid] = v; __syncthreads();
    for (int s = blockDim.x >> 1; s > 0; s >>= 1) {
        if (tid < s) sb[tid] = fmaxf(sb[tid], sb[tid + s]);
        __syncthreads();
    }
    float r = sb[0]; __syncthreads();
    return r;
}

// --------------------------- unified WMMA GEMM ----------------------------
// C[z][m,n] = act( scale * sum_k A[z][m,k] * B[z][k,n] + bias[n] )
//   A: f16 row-major (lda), B: f16 K-major (ldb), C: f32 (Cf) or f16 (Ch).
// Batch z decomposes zb = z/nh, zh = z%nh with separate strides (per-head
// column offsets inside packed QKV buffers).
// Each wave computes a 32x32 C tile: 4 wmma per 32-wide K step, operands
// loaded as: A 16-byte vectors (global_load_b128), B strided 2-byte elements, zero conversions in loop.
__global__ void wmma_gemm_kernel(
    const _Float16* __restrict__ A, const _Float16* __restrict__ B,
    const float* __restrict__ bias, float* __restrict__ Cf,
    _Float16* __restrict__ Ch,
    int M, int N, int K, int lda, int ldb, int ldc,
    long long a_bs, long long a_hs, long long b_bs, long long b_hs,
    long long c_bs, long long c_hs, int nh, float scale, int act)
{
    int z  = blockIdx.z;
    int zb = z / nh;
    int zh = z - zb * nh;
    const _Float16* Ab = A + zb * a_bs + zh * a_hs;
    const _Float16* Bb = B + zb * b_bs + zh * b_hs;

    int wave = threadIdx.x >> 5;
    int lane = threadIdx.x & 31;
    int tilesN = N >> 5;
    int tilesM = M >> 5;
    int tile = blockIdx.x * (blockDim.x >> 5) + wave;
    if (tile >= tilesM * tilesN) return;       // wave-uniform exit (EXEC all-1 for WMMA)
    int tm = (tile / tilesN) << 5;
    int tn = (tile - (tile / tilesN) * tilesN) << 5;

    int mrow = lane & 15;        // A row within tile; also C column
    int half = lane >> 4;        // selects K sub-range (A) / M offset (C)

    v8f acc00 = {0.f,0.f,0.f,0.f,0.f,0.f,0.f,0.f};
    v8f acc01 = acc00, acc10 = acc00, acc11 = acc00;

    const _Float16* aP = Ab + (long long)(tm + mrow) * lda + half * 8;
    const _Float16* bP = Bb + (long long)(half * 8) * ldb + tn + mrow;
    long long bStep = 32LL * ldb;

    union U16 { v16h v; v8h h[2]; };

    for (int k0 = 0; k0 < K; k0 += 32) {
        U16 af0, af1, bf0, bf1;
        // A fragments (16x32 f16): elems 0..7 -> K=k0+half*8+e, 8..15 -> +16
        af0.h[0] = *(const v8h*)(aP);
        af0.h[1] = *(const v8h*)(aP + 16);
        af1.h[0] = *(const v8h*)(aP + 16LL * lda);
        af1.h[1] = *(const v8h*)(aP + 16LL * lda + 16);
        // B fragments (32x16 f16): lane holds N=mrow, elems 0..7 -> K=k0+half*8+e, 8..15 -> +16
        for (int e = 0; e < 8; ++e) {
            bf0.v[e]     = bP[(long long)e * ldb];
            bf0.v[e + 8] = bP[(long long)(e + 16) * ldb];
            bf1.v[e]     = bP[(long long)e * ldb + 16];
            bf1.v[e + 8] = bP[(long long)(e + 16) * ldb + 16];
        }

        __builtin_prefetch(aP + 32, 0, 1);          // global_prefetch_b8
        __builtin_prefetch(bP + bStep, 0, 1);

        acc00 = __builtin_amdgcn_wmma_f32_16x16x32_f16(false, af0.v, false, bf0.v, (short)0, acc00, false, false);
        acc01 = __builtin_amdgcn_wmma_f32_16x16x32_f16(false, af0.v, false, bf1.v, (short)0, acc01, false, false);
        acc10 = __builtin_amdgcn_wmma_f32_16x16x32_f16(false, af1.v, false, bf0.v, (short)0, acc10, false, false);
        acc11 = __builtin_amdgcn_wmma_f32_16x16x32_f16(false, af1.v, false, bf1.v, (short)0, acc11, false, false);

        aP += 32;
        bP += bStep;
    }

    float* CfB = Cf ? Cf + zb * c_bs + zh * c_hs : nullptr;
    _Float16* ChB = Ch ? Ch + zb * c_bs + zh * c_hs : nullptr;

    auto store_tile = [&](v8f acc, int rbase, int cbase) {
        float bv = bias ? bias[cbase] : 0.f;
#pragma unroll
        for (int r = 0; r < 8; ++r) {
            float v = acc[r] * scale + bv;
            if (act == 1) v = 0.5f * v * (1.f + erff(v * 0.70710678118654752f));
            long long off = (long long)(rbase + r) * ldc + cbase;
            if (CfB) CfB[off] = v;
            else     ChB[off] = (_Float16)v;
        }
    };
    // C fragment: acc[r] = C[tile_m + r + 8*half, tile_n + mrow]
    store_tile(acc00, tm + half * 8,      tn + mrow);
    store_tile(acc01, tm + half * 8,      tn + 16 + mrow);
    store_tile(acc10, tm + 16 + half * 8, tn + mrow);
    store_tile(acc11, tm + 16 + half * 8, tn + 16 + mrow);
}

static inline void gemm(hipStream_t st, const _Float16* A, const _Float16* B,
                        const float* bias, float* Cf, _Float16* Ch,
                        int M, int N, int K, int lda, int ldb, int ldc,
                        long long a_bs, long long a_hs, long long b_bs, long long b_hs,
                        long long c_bs, long long c_hs,
                        int nh, int nz, float scale, int act)
{
    int tiles = (M / 32) * (N / 32);
    dim3 grid((tiles + 7) / 8, 1, nz);
    wmma_gemm_kernel<<<grid, 256, 0, st>>>(A, B, bias, Cf, Ch, M, N, K, lda, ldb, ldc,
                                           a_bs, a_hs, b_bs, b_hs, c_bs, c_hs,
                                           nh, scale, act);
}

// --------------------------- conversion kernels ---------------------------
__global__ void cvt_f16_kernel(const float* __restrict__ in,
                               _Float16* __restrict__ out, long long n)
{
    long long i = (long long)blockIdx.x * blockDim.x + threadIdx.x;
    if (i < n) out[i] = (_Float16)in[i];
}

// Wt[k*N + n] = (f16) W[n*K + k]    (weights (N,K) row-major -> K-major f16)
__global__ void transpose_cvt_kernel(const float* __restrict__ W,
                                     _Float16* __restrict__ Wt, int N, int K)
{
    long long i = (long long)blockIdx.x * blockDim.x + threadIdx.x;
    if (i >= (long long)N * K) return;
    int k = (int)(i % K);
    long long n = i / K;
    Wt[(long long)k * N + n] = (_Float16)W[i];
}

// Per-head transpose of K-matrix: dst[(z*DH + c)*Tk + t] = src[zb*src_bs + t*src_rs + zh*DH + c]
__global__ void headT_kernel(const _Float16* __restrict__ src,
                             _Float16* __restrict__ dst,
                             int Tk, int src_rs, long long src_bs, int nh)
{
    int z = blockIdx.y;
    int zb = z / nh, zh = z - zb * nh;
    long long i = (long long)blockIdx.x * blockDim.x + threadIdx.x;
    if (i >= (long long)Tk * DHEAD) return;
    int c = (int)(i & (DHEAD - 1));
    int t = (int)(i >> 6);
    dst[((long long)z * DHEAD + c) * Tk + t] =
        src[zb * src_bs + (long long)t * src_rs + zh * DHEAD + c];
}

// ------------------------------ aux kernels -------------------------------
__global__ void embed_kernel(const int* __restrict__ ids,
                             const float* __restrict__ temb,
                             const float* __restrict__ pemb,
                             float* __restrict__ tok, float* __restrict__ x,
                             _Float16* __restrict__ xh)
{
    long long i = (long long)blockIdx.x * blockDim.x + threadIdx.x;  // BT*D
    int d  = (int)(i & (DMODEL - 1));
    long long bt = i >> 9;
    int t = (int)(bt & (TLEN - 1));
    float tv = temb[(long long)ids[bt] * DMODEL + d];
    float xv = tv + pemb[(long long)t * DMODEL + d];
    tok[i] = tv;
    x[i]   = xv;
    xh[i]  = (_Float16)xv;
}

// rows = nz*Tq; causal mask + softmax in place (f32) and f16 copy for AV gemm.
__global__ void softmax_kernel(float* __restrict__ p, _Float16* __restrict__ ph,
                               int Tk, int Tq, int causal)
{
    __shared__ float sb[256];
    long long row = blockIdx.x;
    float* r = p + row * Tk;
    _Float16* rh = ph + row * Tk;
    int q = (int)(row % Tq);
    float mx = -3.0e38f;
    for (int i = threadIdx.x; i < Tk; i += blockDim.x) {
        float v = r[i];
        if (causal && i > q) { v = -1e30f; r[i] = v; }
        mx = fmaxf(mx, v);
    }
    mx = block_max(mx, sb);
    float se = 0.f;
    for (int i = threadIdx.x; i < Tk; i += blockDim.x) se += expf(r[i] - mx);
    se = block_sum(se, sb);
    float inv = 1.f / se;
    for (int i = threadIdx.x; i < Tk; i += blockDim.x) {
        float v = expf(r[i] - mx) * inv;
        r[i]  = v;
        rh[i] = (_Float16)v;
    }
}

// x = LayerNorm(x + res) * g + b ; also writes f16 copy xh.
__global__ void add_ln_kernel(float* __restrict__ x, const float* __restrict__ res,
                              const float* __restrict__ g, const float* __restrict__ b,
                              _Float16* __restrict__ xh)
{
    __shared__ float sb[256];
    long long row = blockIdx.x;
    int t = threadIdx.x;
    float v0 = x[row * DMODEL + t]       + res[row * DMODEL + t];
    float v1 = x[row * DMODEL + t + 256] + res[row * DMODEL + t + 256];
    float m = block_sum(v0 + v1, sb) * (1.f / DMODEL);
    float d0 = v0 - m, d1 = v1 - m;
    float var = block_sum(d0 * d0 + d1 * d1, sb) * (1.f / DMODEL);
    float inv = rsqrtf(var + 1e-5f);
    float o0 = d0 * inv * g[t]       + b[t];
    float o1 = d1 * inv * g[t + 256] + b[t + 256];
    x[row * DMODEL + t]        = o0;
    x[row * DMODEL + t + 256]  = o1;
    xh[row * DMODEL + t]       = (_Float16)o0;
    xh[row * DMODEL + t + 256] = (_Float16)o1;
}

// pc[b,t,s] = normalize_s( mean_h(cross_probs[b,h,t,s]) * schema_mask[b,s] )
__global__ void pc_kernel(const float* __restrict__ cprobs,
                          const unsigned char* __restrict__ smask,
                          float* __restrict__ pcb, _Float16* __restrict__ pch)
{
    __shared__ float sb[256];
    int bt = blockIdx.x;
    int b = bt >> 8, t = bt & (TLEN - 1);
    float psum = 0.f;
    for (int s = threadIdx.x; s < SLEN; s += blockDim.x) {
        float v = 0.f;
#pragma unroll
        for (int h = 0; h < NHEAD; ++h)
            v += cprobs[(((long long)(b * NHEAD + h) * TLEN + t) * SLEN) + s];
        v = (v * (1.f / NHEAD)) * (smask[b * SLEN + s] ? 1.f : 0.f);
        pcb[(long long)bt * SLEN + s] = v;
        psum += v;
    }
    float tot = block_sum(psum, sb);
    float inv = 1.f / (tot + 1e-8f);
    for (int s = threadIdx.x; s < SLEN; s += blockDim.x) {
        float v = pcb[(long long)bt * SLEN + s] * inv;
        pcb[(long long)bt * SLEN + s] = v;
        pch[(long long)bt * SLEN + s] = (_Float16)v;
    }
}

// p_gen[bt] = sigmoid( [x, c_t, tok] . pgen_w + pgen_b )
__global__ void pgen_kernel(const float* __restrict__ x, const float* __restrict__ ct,
                            const float* __restrict__ tok, const float* __restrict__ pw,
                            const float* __restrict__ pb, float* __restrict__ pg)
{
    __shared__ float sb[256];
    int bt = blockIdx.x;
    float acc = 0.f;
    for (int d = threadIdx.x; d < DMODEL; d += blockDim.x) {
        long long o = (long long)bt * DMODEL + d;
        acc += x[o] * pw[d] + ct[o] * pw[DMODEL + d] + tok[o] * pw[2 * DMODEL + d];
    }
    float s = block_sum(acc, sb);
    if (threadIdx.x == 0) pg[bt] = 1.f / (1.f + expf(-(s + pb[0])));
}

// d_out row (logits) -> p_gen * softmax(logits)
__global__ void vocab_softmax_kernel(float* __restrict__ out, const float* __restrict__ pg)
{
    __shared__ float sb[256];
    int bt = blockIdx.x;
    float* row = out + (long long)bt * VOCAB;
    float mx = -3.0e38f;
    for (int v = threadIdx.x; v < VOCAB; v += blockDim.x) mx = fmaxf(mx, row[v]);
    mx = block_max(mx, sb);
    float se = 0.f;
    for (int v = threadIdx.x; v < VOCAB; v += blockDim.x) se += expf(row[v] - mx);
    se = block_sum(se, sb);
    float inv = pg[bt] / se;
    for (int v = threadIdx.x; v < VOCAB; v += blockDim.x)
        row[v] = expf(row[v] - mx) * inv;
}

// d_out[b,t, src_ids[b,s]] += (1-p_gen[bt]) * pc[b,t,s]
__global__ void scatter_kernel(float* __restrict__ out, const int* __restrict__ src_ids,
                               const float* __restrict__ pcb, const float* __restrict__ pg)
{
    int bt = blockIdx.x;
    int s  = threadIdx.x;                  // block = 512
    int b  = bt >> 8;
    float w = (1.f - pg[bt]) * pcb[(long long)bt * SLEN + s];
    atomicAdd(out + (long long)bt * VOCAB + src_ids[b * SLEN + s], w);
}

__global__ void log_kernel(float* __restrict__ out, long long n)
{
    long long i = (long long)blockIdx.x * blockDim.x + threadIdx.x;
    if (i < n) out[i] = logf(out[i] + 1e-9f);
}

// ------------------------------- launcher ---------------------------------
extern "C" void kernel_launch(void* const* d_in, const int* in_sizes, int n_in,
                              void* d_out, int out_size, void* d_ws, size_t ws_size,
                              hipStream_t stream)
{
    (void)in_sizes; (void)n_in; (void)out_size; (void)ws_size;

    const int*   tgt    = (const int*)d_in[0];
    const int*   src    = (const int*)d_in[1];
    const float* memory = (const float*)d_in[2];
    const unsigned char* smask = (const unsigned char*)d_in[3];   // jnp.bool_ -> 1 byte
    const float* temb   = (const float*)d_in[4];
    const float* pemb   = (const float*)d_in[5];
    const float* lmh    = (const float*)d_in[6];
    const float* pgw    = (const float*)d_in[7];
    const float* pgb    = (const float*)d_in[8];
    const float* sa_wqkv = (const float*)d_in[9];
    const float* sa_bqkv = (const float*)d_in[10];
    const float* sa_wo   = (const float*)d_in[11];
    const float* sa_bo   = (const float*)d_in[12];
    const float* ca_wqkv = (const float*)d_in[13];
    const float* ca_bqkv = (const float*)d_in[14];
    const float* ca_wo   = (const float*)d_in[15];
    const float* ca_bo   = (const float*)d_in[16];
    const float* w1 = (const float*)d_in[17];
    const float* b1 = (const float*)d_in[18];
    const float* w2 = (const float*)d_in[19];
    const float* b2 = (const float*)d_in[20];
    const float* ln1g = (const float*)d_in[21];
    const float* ln1b = (const float*)d_in[22];
    const float* ln2g = (const float*)d_in[23];
    const float* ln2b = (const float*)d_in[24];
    const float* ln3g = (const float*)d_in[25];
    const float* ln3b = (const float*)d_in[26];

    // ---- workspace layout: f32 region then f16 region (~107 MB total) ----
    float* x      = (float*)d_ws;                               // BT*D
    float* tok    = x      + (long long)BT * DMODEL;
    float* proj   = tok    + (long long)BT * DMODEL;            // BT*D
    float* sprobs = proj   + (long long)BT * DMODEL;            // B*H*T*T
    float* cprobs = sprobs + (long long)BATCH * NHEAD * TLEN * TLEN;  // B*H*T*S
    float* pcb    = cprobs + (long long)BATCH * NHEAD * TLEN * SLEN;  // B*T*S
    float* ct     = pcb    + (long long)BT * SLEN;              // BT*D
    float* pg     = ct     + (long long)BT * DMODEL;            // BT

    _Float16* xh     = (_Float16*)(pg + 1024);
    _Float16* qkvh   = xh     + (long long)BT * DMODEL;          // BT*3D
    _Float16* kTs    = qkvh   + (long long)BT * 3 * DMODEL;      // B*H*DH*T
    _Float16* cqh    = kTs    + (long long)BATCH * NHEAD * DHEAD * TLEN;   // BT*D
    _Float16* ckvh   = cqh    + (long long)BT * DMODEL;          // BS*2D
    _Float16* ckvT   = ckvh   + (long long)BS * 2 * DMODEL;      // B*H*DH*S
    _Float16* attH   = ckvT   + (long long)BATCH * NHEAD * DHEAD * SLEN;   // BT*D
    _Float16* ffh    = attH   + (long long)BT * DMODEL;          // BT*DFF
    _Float16* sprbH  = ffh    + (long long)BT * DFF;             // B*H*T*T
    _Float16* cprbH  = sprbH  + (long long)BATCH * NHEAD * TLEN * TLEN;    // B*H*T*S
    _Float16* pch    = cprbH  + (long long)BATCH * NHEAD * TLEN * SLEN;    // B*T*S
    _Float16* memh   = pch    + (long long)BT * SLEN;            // BS*D
    _Float16* lmhT   = memh   + (long long)BS * DMODEL;          // D*V
    _Float16* wqkvT  = lmhT   + (long long)DMODEL * VOCAB;       // D*3D
    _Float16* woT    = wqkvT  + (long long)DMODEL * 3 * DMODEL;  // D*D
    _Float16* w1T    = woT    + (long long)DMODEL * DMODEL;      // D*DFF
    _Float16* w2T    = w1T    + (long long)DMODEL * DFF;         // DFF*D
    float* outF = (float*)d_out;

    // 1) one-time conversions
    embed_kernel<<<(BT * DMODEL) / 256, 256, 0, stream>>>(tgt, temb, pemb, tok, x, xh);
    cvt_f16_kernel<<<(BS * DMODEL) / 256, 256, 0, stream>>>(memory, memh, (long long)BS * DMODEL);
    transpose_cvt_kernel<<<((long long)VOCAB * DMODEL + 255) / 256, 256, 0, stream>>>(lmh, lmhT, VOCAB, DMODEL);

    for (int l = 0; l < NLAYER; ++l) {
        const float* Wq = sa_wqkv + (long long)l * 3 * DMODEL * DMODEL;
        const float* Bq = sa_bqkv + (long long)l * 3 * DMODEL;
        const float* Wc = ca_wqkv + (long long)l * 3 * DMODEL * DMODEL;
        const float* Bc = ca_bqkv + (long long)l * 3 * DMODEL;

        // --- self-attention ---
        transpose_cvt_kernel<<<(3 * DMODEL * DMODEL) / 256, 256, 0, stream>>>(Wq, wqkvT, 3 * DMODEL, DMODEL);
        gemm(stream, xh, wqkvT, Bq, nullptr, qkvh, BT, 3 * DMODEL, DMODEL,
             DMODEL, 3 * DMODEL, 3 * DMODEL, 0, 0, 0, 0, 0, 0, 1, 1, 1.f, 0);
        headT_kernel<<<dim3((TLEN * DHEAD) / 256, BATCH * NHEAD), 256, 0, stream>>>(
            qkvh + DMODEL, kTs, TLEN, 3 * DMODEL, (long long)TLEN * 3 * DMODEL, NHEAD);
        // scores = Q.K^T / 8  (z = b*H+h)
        gemm(stream, qkvh, kTs, nullptr, sprobs, nullptr, TLEN, TLEN, DHEAD,
             3 * DMODEL, TLEN, TLEN,
             (long long)TLEN * 3 * DMODEL, DHEAD,
             (long long)NHEAD * DHEAD * TLEN, (long long)DHEAD * TLEN,
             (long long)NHEAD * TLEN * TLEN, (long long)TLEN * TLEN,
             NHEAD, BATCH * NHEAD, 0.125f, 0);
        softmax_kernel<<<BATCH * NHEAD * TLEN, 256, 0, stream>>>(sprobs, sprbH, TLEN, TLEN, 1);
        // att = P @ V
        gemm(stream, sprbH, qkvh + 2 * DMODEL, nullptr, nullptr, attH, TLEN, DHEAD, TLEN,
             TLEN, 3 * DMODEL, DMODEL,
             (long long)NHEAD * TLEN * TLEN, (long long)TLEN * TLEN,
             (long long)TLEN * 3 * DMODEL, DHEAD,
             (long long)TLEN * DMODEL, DHEAD,
             NHEAD, BATCH * NHEAD, 1.f, 0);
        transpose_cvt_kernel<<<(DMODEL * DMODEL) / 256, 256, 0, stream>>>(
            sa_wo + (long long)l * DMODEL * DMODEL, woT, DMODEL, DMODEL);
        gemm(stream, attH, woT, sa_bo + l * DMODEL, proj, nullptr, BT, DMODEL, DMODEL,
             DMODEL, DMODEL, DMODEL, 0, 0, 0, 0, 0, 0, 1, 1, 1.f, 0);
        add_ln_kernel<<<BT, 256, 0, stream>>>(x, proj, ln1g + l * DMODEL, ln1b + l * DMODEL, xh);

        // --- cross-attention ---
        transpose_cvt_kernel<<<(3 * DMODEL * DMODEL) / 256, 256, 0, stream>>>(Wc, wqkvT, 3 * DMODEL, DMODEL);
        gemm(stream, xh, wqkvT, Bc, nullptr, cqh, BT, DMODEL, DMODEL,
             DMODEL, 3 * DMODEL, DMODEL, 0, 0, 0, 0, 0, 0, 1, 1, 1.f, 0);
        gemm(stream, memh, wqkvT + DMODEL, Bc + DMODEL, nullptr, ckvh, BS, 2 * DMODEL, DMODEL,
             DMODEL, 3 * DMODEL, 2 * DMODEL, 0, 0, 0, 0, 0, 0, 1, 1, 1.f, 0);
        headT_kernel<<<dim3((SLEN * DHEAD) / 256, BATCH * NHEAD), 256, 0, stream>>>(
            ckvh, ckvT, SLEN, 2 * DMODEL, (long long)SLEN * 2 * DMODEL, NHEAD);
        gemm(stream, cqh, ckvT, nullptr, cprobs, nullptr, TLEN, SLEN, DHEAD,
             DMODEL, SLEN, SLEN,
             (long long)TLEN * DMODEL, DHEAD,
             (long long)NHEAD * DHEAD * SLEN, (long long)DHEAD * SLEN,
             (long long)NHEAD * TLEN * SLEN, (long long)TLEN * SLEN,
             NHEAD, BATCH * NHEAD, 0.125f, 0);
        softmax_kernel<<<BATCH * NHEAD * TLEN, 256, 0, stream>>>(cprobs, cprbH, SLEN, TLEN, 0);
        gemm(stream, cprbH, ckvh + DMODEL, nullptr, nullptr, attH, TLEN, DHEAD, SLEN,
             SLEN, 2 * DMODEL, DMODEL,
             (long long)NHEAD * TLEN * SLEN, (long long)TLEN * SLEN,
             (long long)SLEN * 2 * DMODEL, DHEAD,
             (long long)TLEN * DMODEL, DHEAD,
             NHEAD, BATCH * NHEAD, 1.f, 0);
        transpose_cvt_kernel<<<(DMODEL * DMODEL) / 256, 256, 0, stream>>>(
            ca_wo + (long long)l * DMODEL * DMODEL, woT, DMODEL, DMODEL);
        gemm(stream, attH, woT, ca_bo + l * DMODEL, proj, nullptr, BT, DMODEL, DMODEL,
             DMODEL, DMODEL, DMODEL, 0, 0, 0, 0, 0, 0, 1, 1, 1.f, 0);
        add_ln_kernel<<<BT, 256, 0, stream>>>(x, proj, ln2g + l * DMODEL, ln2b + l * DMODEL, xh);

        // --- FFN ---
        transpose_cvt_kernel<<<(DFF * DMODEL) / 256, 256, 0, stream>>>(
            w1 + (long long)l * DFF * DMODEL, w1T, DFF, DMODEL);
        transpose_cvt_kernel<<<(DMODEL * DFF) / 256, 256, 0, stream>>>(
            w2 + (long long)l * DMODEL * DFF, w2T, DMODEL, DFF);
        gemm(stream, xh, w1T, b1 + (long long)l * DFF, nullptr, ffh, BT, DFF, DMODEL,
             DMODEL, DFF, DFF, 0, 0, 0, 0, 0, 0, 1, 1, 1.f, 1);
        gemm(stream, ffh, w2T, b2 + (long long)l * DMODEL, proj, nullptr, BT, DMODEL, DFF,
             DFF, DMODEL, DMODEL, 0, 0, 0, 0, 0, 0, 1, 1, 1.f, 0);
        add_ln_kernel<<<BT, 256, 0, stream>>>(x, proj, ln3g + l * DMODEL, ln3b + l * DMODEL, xh);
    }

    // 2) copy distribution from last layer's cross-attn probs
    pc_kernel<<<BT, 256, 0, stream>>>(cprobs, smask, pcb, pch);

    // 3) c_t = pc @ memory   (z = batch; memory already K-major)
    gemm(stream, pch, memh, nullptr, ct, nullptr, TLEN, DMODEL, SLEN,
         SLEN, DMODEL, DMODEL,
         (long long)TLEN * SLEN, 0, (long long)SLEN * DMODEL, 0,
         (long long)TLEN * DMODEL, 0,
         1, BATCH, 1.f, 0);

    // 4) gate
    pgen_kernel<<<BT, 256, 0, stream>>>(x, ct, tok, pgw, pgb, pg);

    // 5) logits = x @ lm_head^T -> d_out (scratch)
    gemm(stream, xh, lmhT, nullptr, outF, nullptr, BT, VOCAB, DMODEL,
         DMODEL, VOCAB, VOCAB, 0, 0, 0, 0, 0, 0, 1, 1, 1.f, 0);

    // 6) p_gen * softmax; scatter copy probs; log
    vocab_softmax_kernel<<<BT, 256, 0, stream>>>(outF, pg);
    scatter_kernel<<<BT, SLEN, 0, stream>>>(outF, src, pcb, pg);
    long long n = (long long)BT * VOCAB;
    log_kernel<<<(unsigned)((n + 255) / 256), 256, 0, stream>>>(outF, n);
}
